// GPTBlock_2499670966566
// MI455X (gfx1250) — compile-verified, matrix operand order checked by reading, not run on hardware
//
#include <hip/hip_runtime.h>
#include <math.h>

// ---------------------------------------------------------------------------
// Problem constants (from reference)
// ---------------------------------------------------------------------------
#define BB 4
#define TT 2048
#define MM 1024
#define HH 16
#define DD 64
#define EE 8
#define SS (BB*TT)          // 8192 tokens
#define CC (SS/EE)          // 1024 capacity
#define FF (4*MM)           // 4096 ffn hidden

typedef __attribute__((ext_vector_type(16))) __bf16 v16bf;
typedef __attribute__((ext_vector_type(8)))  __bf16 v8bf;
typedef __attribute__((ext_vector_type(8)))  float  v8f;
typedef __attribute__((ext_vector_type(4)))  unsigned int v4u;
typedef __attribute__((ext_vector_type(8)))  int v8i;
typedef __attribute__((ext_vector_type(4)))  int v4i;

// Compiler-level memory clobber (no instructions emitted).
__device__ __forceinline__ void tdm_mem_publish() { asm volatile("" ::: "memory"); }

// Escape an LDS pointer into an asm black box with a memory clobber. After
// this, alias analysis must assume external writers (the TDM DMA engine) can
// modify the pointed-to LDS, so loads from it are never folded to undef and
// never store-forwarded across the publish clobbers.
__device__ __forceinline__ void capture_lds(void* p) {
    asm volatile("" : "+v"(p) :: "memory");
}

// Build a 16x32 bf16 WMMA fragment (A or B operand) from an LDS row of 32
// contiguous bf16 values. Per ISA: lanes 0-15 hold K in {klo..klo+7} and
// {klo+16..klo+23} with klo = (lane>=16 ? 8 : 0).
__device__ __forceinline__ v16bf load_frag(const __bf16* p, int klo) {
    const v8bf* q = (const v8bf*)p;
    v8bf lo = q[klo >> 3];
    v8bf hi = q[(klo >> 3) + 2];
    return __builtin_shufflevector(lo, hi, 0,1,2,3,4,5,6,7,8,9,10,11,12,13,14,15);
}

__device__ __forceinline__ float gelu_exact(float x) {
    return 0.5f * x * (1.0f + erff(x * 0.70710678118654752f));
}

// ---------------------------------------------------------------------------
// TDM: post a 2D bf16 tile load (tile_d1 rows x tile_d0 elems, row stride
// stride0 elems) from global into LDS at byte offset lds_byte. Tracked with
// TENSORcnt. Issue from ONE wave only.
// ---------------------------------------------------------------------------
__device__ __forceinline__ void tdm_load_2d(unsigned lds_byte, const void* gptr,
                                            unsigned tile_d0, unsigned tile_d1,
                                            unsigned long long stride0_elems)
{
    unsigned long long ga = (unsigned long long)gptr;
    v4u g0;
    g0.x = 1u;                                           // count=1, user descriptor
    g0.y = lds_byte;                                     // lds_addr
    g0.z = (unsigned)(ga & 0xffffffffull);               // global_addr[31:0]
    g0.w = (unsigned)((ga >> 32) & 0x1ffffffull) | (2u << 30);  // addr[56:32] | type=2
    v8i g1;
    g1[0] = (int)(1u << 16);                             // data_size = 1 -> 2 bytes
    g1[1] = (int)((tile_d0 & 0xffffu) << 16);            // tensor_dim0[15:0]
    g1[2] = (int)((tile_d0 >> 16) | ((tile_d1 & 0xffffu) << 16)); // td0 hi | tensor_dim1 lo
    g1[3] = (int)((tile_d1 >> 16) | (tile_d0 << 16));    // td1 hi | tile_dim0
    g1[4] = (int)(tile_d1 & 0xffffu);                    // tile_dim1 (tile_dim2=0)
    g1[5] = (int)(stride0_elems & 0xffffffffull);        // tensor_dim0_stride lo
    g1[6] = (int)((stride0_elems >> 32) & 0xffffull);    // stride hi (dim1_stride lo = 0)
    g1[7] = 0;
    v4i g2 = {0,0,0,0};
    v4i g3 = {0,0,0,0};
#if defined(__clang_major__) && (__clang_major__ >= 23)
    v8i g4 = {0,0,0,0,0,0,0,0};
    __builtin_amdgcn_tensor_load_to_lds(g0, g1, g2, g3, g4, 0);
#else
    __builtin_amdgcn_tensor_load_to_lds(g0, g1, g2, g3, 0);
#endif
    tdm_mem_publish();   // model the DMA as a writer at this program point
}

// ---------------------------------------------------------------------------
// fp32 -> bf16 elementwise convert (n multiple of 1024)
// ---------------------------------------------------------------------------
__global__ __launch_bounds__(256) void cvt_kernel(const float* __restrict__ in,
                                                  __bf16* __restrict__ o) {
    size_t i = ((size_t)blockIdx.x * 256 + threadIdx.x) * 4;
    float4 f = *(const float4*)(in + i);
    o[i+0] = (__bf16)f.x; o[i+1] = (__bf16)f.y;
    o[i+2] = (__bf16)f.z; o[i+3] = (__bf16)f.w;
}

// ---------------------------------------------------------------------------
// fp32 (R x Cc, batched) -> bf16 transposed (Cc x R): out[c*R+r] = in[r*Cc+c]
// ---------------------------------------------------------------------------
__global__ __launch_bounds__(256) void transpose_cvt_kernel(
    const float* __restrict__ in, __bf16* __restrict__ o, int R, int Cc)
{
    __shared__ float t[32][33];
    int e = blockIdx.z;
    const float* ine = in + (size_t)e * R * Cc;
    __bf16* oe = o + (size_t)e * R * Cc;
    int c0 = blockIdx.x * 32, r0 = blockIdx.y * 32;
    int tid = threadIdx.x;
    int r = tid >> 3, c = (tid & 7) * 4;
    float4 f = *(const float4*)(ine + (size_t)(r0 + r) * Cc + c0 + c);
    t[r][c+0] = f.x; t[r][c+1] = f.y; t[r][c+2] = f.z; t[r][c+3] = f.w;
    __syncthreads();
    int cc = tid >> 3, rr = (tid & 7) * 4;
    __bf16* orow = oe + (size_t)(c0 + cc) * R + r0;
    orow[rr+0] = (__bf16)t[rr+0][cc]; orow[rr+1] = (__bf16)t[rr+1][cc];
    orow[rr+2] = (__bf16)t[rr+2][cc]; orow[rr+3] = (__bf16)t[rr+3][cc];
}

// ---------------------------------------------------------------------------
// LayerNorm: one block per token row of 1024; fp32 in, bf16 out
// ---------------------------------------------------------------------------
__global__ __launch_bounds__(256) void ln_kernel(const float* __restrict__ x,
                                                 const float* __restrict__ g,
                                                 const float* __restrict__ b,
                                                 __bf16* __restrict__ o) {
    __shared__ float ssum[256], ssq[256];
    int s = blockIdx.x, tid = threadIdx.x;
    const float4* row = (const float4*)(x + (size_t)s * MM);
    float4 f = row[tid];
    ssum[tid] = f.x + f.y + f.z + f.w;
    ssq[tid]  = f.x*f.x + f.y*f.y + f.z*f.z + f.w*f.w;
    __syncthreads();
    for (int st = 128; st > 0; st >>= 1) {
        if (tid < st) { ssum[tid] += ssum[tid+st]; ssq[tid] += ssq[tid+st]; }
        __syncthreads();
    }
    float mu  = ssum[0] * (1.0f / MM);
    float var = ssq[0] * (1.0f / MM) - mu * mu;
    float rs  = rsqrtf(var + 1e-5f);
    float4 gg = ((const float4*)g)[tid];
    float4 bb = ((const float4*)b)[tid];
    __bf16* orow = o + (size_t)s * MM + tid * 4;
    orow[0] = (__bf16)((f.x - mu) * rs * gg.x + bb.x);
    orow[1] = (__bf16)((f.y - mu) * rs * gg.y + bb.y);
    orow[2] = (__bf16)((f.z - mu) * rs * gg.z + bb.z);
    orow[3] = (__bf16)((f.w - mu) * rs * gg.w + bb.w);
}

// ---------------------------------------------------------------------------
// bf16 WMMA GEMM, TDM-fed, LDS double-buffered. Computes out = epi(A @ W^T).
//   A: rows x K bf16 row-major, W: N x K bf16 row-major (pre-transposed).
//   obf: 1 -> bf16 output, 0 -> f32 output. act: 1 -> exact GELU.
//   bias/res optional. blockIdx.z batches experts via element strides.
// Per chunk: wave0 posts next A/B tile TDMs -> all waves WMMA on current LDS
// buffer -> wave0 s_wait_tensorcnt 0 -> one barrier.
// ---------------------------------------------------------------------------
__global__ __launch_bounds__(256) void gemm_kernel(
    const __bf16* __restrict__ A, const __bf16* __restrict__ W,
    const float* __restrict__ bias, const float* __restrict__ res,
    void* __restrict__ out, int obf,
    int N, int K, int act,
    long long sA, long long sW, long long sBias, long long sOut)
{
    // [0 .. 8192)  : As  2 x 128 x 32
    // [8192..16384): Bs  2 x 128 x 32
    __shared__ __bf16 smem[16384];

    int e = blockIdx.z;
    const __bf16* Ae = A + (size_t)e * sA;
    const __bf16* We = W + (size_t)e * sW;
    const float* Be = bias ? bias + (size_t)e * sBias : nullptr;

    int n0 = blockIdx.x * 128, r0 = blockIdx.y * 128;
    int tid = threadIdx.x;
    int wave = tid >> 5, lane = tid & 31, l16 = lane & 15, half = lane >> 4;
    int klo = half * 8;

    // Escape the LDS address (TDM is an invisible writer) and give the LDS
    // variable real stores so its loads are never folded to undef.
    capture_lds((void*)smem);
    {
        v8bf z;
#pragma unroll
        for (int j = 0; j < 8; j++) z[j] = (__bf16)0.0f;
        for (int i = tid; i < 2048; i += 256) ((v8bf*)smem)[i] = z;
    }
    __syncthreads();   // zero-init complete before the DMA writes

    auto issue_tdm = [&](int k0, int buf) {
        tdm_load_2d((unsigned)(buf * 8192),         Ae + (size_t)r0 * K + k0, 32u, 128u, (unsigned long long)K);
        tdm_load_2d((unsigned)(16384 + buf * 8192), We + (size_t)n0 * K + k0, 32u, 128u, (unsigned long long)K);
    };

    v8f acc[8];
#pragma unroll
    for (int i = 0; i < 8; i++)
        for (int j = 0; j < 8; j++) acc[i][j] = 0.0f;

    if (wave == 0) {
        issue_tdm(0, 0);
        __builtin_amdgcn_s_wait_tensorcnt(0);
    }
    __syncthreads();
    tdm_mem_publish();     // LDS now holds DMA data

    int nchunks = K >> 5;
    for (int c = 0; c < nchunks; ++c) {
        int cur = c & 1;
        bool more = (c + 1) < nchunks;
        if (more && wave == 0) issue_tdm((c + 1) << 5, cur ^ 1);  // overlap DMA with WMMA

        v16bf a = load_frag(&smem[cur * 4096 + (wave * 16 + l16) * 32], klo);
#pragma unroll
        for (int g = 0; g < 2; ++g) {
            // batch 4 fragment loads, then 4 back-to-back WMMAs
            v16bf b0 = load_frag(&smem[8192 + cur * 4096 + ((g*4+0)*16 + l16) * 32], klo);
            v16bf b1 = load_frag(&smem[8192 + cur * 4096 + ((g*4+1)*16 + l16) * 32], klo);
            v16bf b2 = load_frag(&smem[8192 + cur * 4096 + ((g*4+2)*16 + l16) * 32], klo);
            v16bf b3 = load_frag(&smem[8192 + cur * 4096 + ((g*4+3)*16 + l16) * 32], klo);
            acc[g*4+0] = __builtin_amdgcn_wmma_f32_16x16x32_bf16(false, a, false, b0, (short)0, acc[g*4+0], false, false);
            acc[g*4+1] = __builtin_amdgcn_wmma_f32_16x16x32_bf16(false, a, false, b1, (short)0, acc[g*4+1], false, false);
            acc[g*4+2] = __builtin_amdgcn_wmma_f32_16x16x32_bf16(false, a, false, b2, (short)0, acc[g*4+2], false, false);
            acc[g*4+3] = __builtin_amdgcn_wmma_f32_16x16x32_bf16(false, a, false, b3, (short)0, acc[g*4+3], false, false);
        }
        if (more && wave == 0) __builtin_amdgcn_s_wait_tensorcnt(0);
        __syncthreads();
        tdm_mem_publish(); // next buffer published by the DMA + barrier
    }

    // epilogue: C layout -> VGPR r holds rows half*8+r, lane%16 = column
    __bf16* Ob = (__bf16*)out + (size_t)e * sOut;
    float*  Of = (float*)out  + (size_t)e * sOut;
#pragma unroll
    for (int nt = 0; nt < 8; nt++) {
#pragma unroll
        for (int r = 0; r < 8; r++) {
            int row = r0 + wave * 16 + half * 8 + r;
            int col = n0 + nt * 16 + l16;
            float v = acc[nt][r];
            if (Be) v += Be[col];
            if (act == 1) v = gelu_exact(v);
            if (res) v += res[(size_t)row * N + col];
            if (obf) Ob[(size_t)row * N + col] = (__bf16)v;
            else     Of[(size_t)row * N + col] = v;
        }
    }
}

// ---------------------------------------------------------------------------
// Flash attention, causal, bf16 QKV. One block = one (b, h, 128-query tile),
// 8 waves x 16 query rows. Q and K tiles via TDM; V staged manually (needs
// transpose). Keys streamed 32 at a time, double-buffered.
// ---------------------------------------------------------------------------
__global__ __launch_bounds__(256) void attn_kernel(const __bf16* __restrict__ qkv,
                                                   __bf16* __restrict__ o_out)
{
    // elems: Qs [0..8192) 128x64 | Ks [8192..12288) 2x32x64
    //        Vst [12288..16384) 2x64x32 | Ps [16384..20480) 8x16x32
    __shared__ __bf16 smem[20480];

    int tid = threadIdx.x;
    int wave = tid >> 5, lane = tid & 31, l16 = lane & 15, half = lane >> 4;
    int klo = half * 8;
    int b = blockIdx.z, h = blockIdx.y, q0 = blockIdx.x * 128;

    capture_lds((void*)smem);
    {
        v8bf z;
#pragma unroll
        for (int j = 0; j < 8; j++) z[j] = (__bf16)0.0f;
        for (int i = tid; i < 2560; i += 256) ((v8bf*)smem)[i] = z;
    }

    // V staging coords: 32 keys x 64 d = 2048 elems, 8 per thread
    int vkey = tid >> 3, vd0 = (tid & 7) * 8;
    v8bf vreg;

    auto load_v = [&](int kbase) {
        vreg = *(const v8bf*)(qkv + ((size_t)(b * TT + kbase + vkey)) * (3*MM) + 2*MM + h * DD + vd0);
    };
    auto store_v = [&](int buf) {
#pragma unroll
        for (int j = 0; j < 8; j++)
            smem[12288 + buf * 2048 + (vd0 + j) * 32 + vkey] = vreg[j];
    };
    auto issue_k_tdm = [&](int kbase, int buf) {
        tdm_load_2d((unsigned)((8192 + buf * 2048) * 2),
                    qkv + ((size_t)(b * TT + kbase)) * (3*MM) + MM + h * DD,
                    64u, 32u, (unsigned long long)(3*MM));
    };

    v8f oacc[4];
#pragma unroll
    for (int i = 0; i < 4; i++) for (int j = 0; j < 8; j++) oacc[i][j] = 0.0f;
    float mrow[8], lrow[8];
#pragma unroll
    for (int r = 0; r < 8; r++) { mrow[r] = -1e30f; lrow[r] = 0.0f; }
    const float scale = 0.125f;   // 1/sqrt(64)

    int kend = q0 + 128;          // causal upper bound for this query tile
    load_v(0);
    __syncthreads();   // zero-init complete before the DMA writes
    if (wave == 0) {
        // Q tile (128 x 64) once + first K tile
        tdm_load_2d(0u, qkv + ((size_t)(b * TT + q0)) * (3*MM) + h * DD,
                    64u, 128u, (unsigned long long)(3*MM));
        issue_k_tdm(0, 0);
    }
    store_v(0);
    if (wave == 0) __builtin_amdgcn_s_wait_tensorcnt(0);
    __syncthreads();
    tdm_mem_publish();     // Q/K tiles now resident

    int c = 0;
    for (int kbv = 0; kbv < kend; kbv += 32, ++c) {
        int cur = c & 1;
        bool more = (kbv + 32) < kend;
        if (more) {
            load_v(kbv + 32);
            if (wave == 0) issue_k_tdm(kbv + 32, cur ^ 1);
        }

        // scores: 16 (q rows) x 32 (keys), K-dim = D = 64 -> 2 chunks of 32
        const __bf16* qrow = &smem[(wave * 16 + l16) * 64];
        const __bf16* kbase0 = &smem[8192 + cur * 2048];
        v16bf aq0 = load_frag(qrow,      klo);
        v16bf aq1 = load_frag(qrow + 32, klo);
        v16bf bk00 = load_frag(kbase0 + l16 * 64,        klo);
        v16bf bk01 = load_frag(kbase0 + l16 * 64 + 32,   klo);
        v16bf bk10 = load_frag(kbase0 + (16 + l16) * 64, klo);
        v16bf bk11 = load_frag(kbase0 + (16 + l16) * 64 + 32, klo);
        v8f sacc[2];
#pragma unroll
        for (int i = 0; i < 2; i++) for (int j = 0; j < 8; j++) sacc[i][j] = 0.0f;
        sacc[0] = __builtin_amdgcn_wmma_f32_16x16x32_bf16(false, aq0, false, bk00, (short)0, sacc[0], false, false);
        sacc[1] = __builtin_amdgcn_wmma_f32_16x16x32_bf16(false, aq0, false, bk10, (short)0, sacc[1], false, false);
        sacc[0] = __builtin_amdgcn_wmma_f32_16x16x32_bf16(false, aq1, false, bk01, (short)0, sacc[0], false, false);
        sacc[1] = __builtin_amdgcn_wmma_f32_16x16x32_bf16(false, aq1, false, bk11, (short)0, sacc[1], false, false);

        // online softmax per query row (row stats replicated across 16 lanes)
#pragma unroll
        for (int r = 0; r < 8; r++) {
            int qg = q0 + wave * 16 + half * 8 + r;
            float v0 = sacc[0][r] * scale; if (kbv + l16 > qg)      v0 = -1e30f;
            float v1 = sacc[1][r] * scale; if (kbv + 16 + l16 > qg) v1 = -1e30f;
            float rm = fmaxf(v0, v1);
            for (int d = 1; d < 16; d <<= 1) rm = fmaxf(rm, __shfl_xor(rm, d, 32));
            float mnew  = fmaxf(mrow[r], rm);
            float alpha = __expf(mrow[r] - mnew);
            mrow[r] = mnew;
            float p0 = __expf(v0 - mnew), p1 = __expf(v1 - mnew);
            float rs = p0 + p1;
            for (int d = 1; d < 16; d <<= 1) rs += __shfl_xor(rs, d, 32);
            lrow[r] = lrow[r] * alpha + rs;
            for (int nt = 0; nt < 4; nt++) oacc[nt][r] *= alpha;
            smem[16384 + wave * 512 + (half * 8 + r) * 32 + l16]      = (__bf16)p0;
            smem[16384 + wave * 512 + (half * 8 + r) * 32 + 16 + l16] = (__bf16)p1;
        }

        // O += P (16x32) @ V (32x64); A-frag from per-wave LDS transpose
        v16bf ap  = load_frag(&smem[16384 + wave * 512 + l16 * 32], klo);
        v16bf bv0 = load_frag(&smem[12288 + cur * 2048 + (0*16 + l16) * 32], klo);
        v16bf bv1 = load_frag(&smem[12288 + cur * 2048 + (1*16 + l16) * 32], klo);
        v16bf bv2 = load_frag(&smem[12288 + cur * 2048 + (2*16 + l16) * 32], klo);
        v16bf bv3 = load_frag(&smem[12288 + cur * 2048 + (3*16 + l16) * 32], klo);
        oacc[0] = __builtin_amdgcn_wmma_f32_16x16x32_bf16(false, ap, false, bv0, (short)0, oacc[0], false, false);
        oacc[1] = __builtin_amdgcn_wmma_f32_16x16x32_bf16(false, ap, false, bv1, (short)0, oacc[1], false, false);
        oacc[2] = __builtin_amdgcn_wmma_f32_16x16x32_bf16(false, ap, false, bv2, (short)0, oacc[2], false, false);
        oacc[3] = __builtin_amdgcn_wmma_f32_16x16x32_bf16(false, ap, false, bv3, (short)0, oacc[3], false, false);

        if (more) {
            store_v(cur ^ 1);
            if (wave == 0) __builtin_amdgcn_s_wait_tensorcnt(0);
        }
        __syncthreads();
        tdm_mem_publish(); // next K tile published
    }

    // normalize and write merged-head output [S, M] bf16
#pragma unroll
    for (int nt = 0; nt < 4; nt++) {
#pragma unroll
        for (int r = 0; r < 8; r++) {
            int qg = q0 + wave * 16 + half * 8 + r;
            int d  = nt * 16 + l16;
            o_out[((size_t)(b * TT + qg)) * MM + h * DD + d] = (__bf16)(oacc[nt][r] / lrow[r]);
        }
    }
}

// ---------------------------------------------------------------------------
// Gating: one wave per token; logits = h2[s] @ wg (wg is [M,E] f32 row-major)
// ---------------------------------------------------------------------------
__global__ __launch_bounds__(256) void gate_kernel(const __bf16* __restrict__ h2,
                                                   const float* __restrict__ wg,
                                                   int* __restrict__ idx,
                                                   float* __restrict__ gv)
{
    int s = blockIdx.x * 8 + (threadIdx.x >> 5);
    int lane = threadIdx.x & 31;
    const __bf16* row = h2 + (size_t)s * MM;
    float acc[8];
    for (int e = 0; e < 8; e++) acc[e] = 0.0f;
    for (int m = lane; m < MM; m += 32) {
        float hv = (float)row[m];
        const float4* wr = (const float4*)(wg + (size_t)m * EE);
        float4 a = wr[0], b = wr[1];
        acc[0] += hv * a.x; acc[1] += hv * a.y; acc[2] += hv * a.z; acc[3] += hv * a.w;
        acc[4] += hv * b.x; acc[5] += hv * b.y; acc[6] += hv * b.z; acc[7] += hv * b.w;
    }
    for (int e = 0; e < 8; e++)
        for (int d = 1; d < 32; d <<= 1) acc[e] += __shfl_xor(acc[e], d, 32);
    if (lane == 0) {
        float mx = acc[0]; int bi = 0;
        for (int e = 1; e < 8; e++) if (acc[e] > mx) { mx = acc[e]; bi = e; }
        float se = 0.0f;
        for (int e = 0; e < 8; e++) se += __expf(acc[e] - mx);
        idx[s] = bi;
        gv[s]  = 1.0f / se;   // softmax value at argmax
    }
}

// ---------------------------------------------------------------------------
// Deterministic per-expert position scan (token order), single block.
// ---------------------------------------------------------------------------
__global__ __launch_bounds__(256) void scan_kernel(const int* __restrict__ idx,
                                                   int* __restrict__ pos)
{
    __shared__ int counts[EE];
    __shared__ int wtot[8][EE];
    int tid = threadIdx.x, wave = tid >> 5, lane = tid & 31;
    if (tid < EE) counts[tid] = 0;
    __syncthreads();
    for (int base = 0; base < SS; base += 256) {
        int my = idx[base + tid];
        unsigned long long lanebit = 1ull << lane;
        int prefix = 0;
        for (int e = 0; e < EE; e++) {
            unsigned long long mask = __ballot(my == e);
            if (lane == 0) wtot[wave][e] = (int)__popcll(mask);
            if (my == e) prefix = (int)__popcll(mask & (lanebit - 1ull));
        }
        __syncthreads();
        int off = counts[my];
        for (int w2 = 0; w2 < wave; w2++) off += wtot[w2][my];
        pos[base + tid] = off + prefix;
        __syncthreads();
        if (tid < EE) {
            int t = 0;
            for (int w2 = 0; w2 < 8; w2++) t += wtot[w2][tid];
            counts[tid] += t;
        }
        __syncthreads();
    }
}

__global__ __launch_bounds__(256) void zero_kernel(float4* __restrict__ p) {
    float4 z; z.x = z.y = z.z = z.w = 0.0f;
    p[(size_t)blockIdx.x * 256 + threadIdx.x] = z;
}

// scatter token row (bf16) -> disp[e][pos] (dropped skipped; disp pre-zeroed)
__global__ __launch_bounds__(256) void scatter_kernel(const __bf16* __restrict__ h2,
                                                      const int* __restrict__ idx,
                                                      const int* __restrict__ pos,
                                                      __bf16* __restrict__ disp)
{
    int s = blockIdx.x;
    int p = pos[s];
    if (p >= CC) return;
    int e = idx[s];
    const uint4* src = (const uint4*)(h2 + (size_t)s * MM);
    uint4* dst = (uint4*)(disp + ((size_t)e * CC + p) * MM);
    if (threadIdx.x < 128) dst[threadIdx.x] = src[threadIdx.x];   // 128 x 8 bf16
}

// out = x1 + gate * eo[idx, pos]  (zero for dropped tokens)
__global__ __launch_bounds__(256) void combine_kernel(const float* __restrict__ x1,
                                                      const float* __restrict__ eo,
                                                      const int* __restrict__ idx,
                                                      const int* __restrict__ pos,
                                                      const float* __restrict__ gv,
                                                      float* __restrict__ out)
{
    int s = blockIdx.x, tid = threadIdx.x;
    int p = pos[s], e = idx[s];
    bool keep = (p < CC);
    float g = keep ? gv[s] : 0.0f;
    const float4* xr = (const float4*)(x1 + (size_t)s * MM);
    const float4* er = (const float4*)(eo + ((size_t)e * CC + (keep ? p : 0)) * MM);
    float4 xv = xr[tid];
    float4 ev = er[tid];
    float4 ov;
    ov.x = xv.x + ev.x * g; ov.y = xv.y + ev.y * g;
    ov.z = xv.z + ev.z * g; ov.w = xv.w + ev.w * g;
    ((float4*)(out + (size_t)s * MM))[tid] = ov;
}

// ---------------------------------------------------------------------------
// Orchestration
// ---------------------------------------------------------------------------
extern "C" void kernel_launch(void* const* d_in, const int* in_sizes, int n_in,
                              void* d_out, int out_size, void* d_ws, size_t ws_size,
                              hipStream_t stream) {
    const float* x     = (const float*)d_in[0];
    const float* ln1_g = (const float*)d_in[1];
    const float* ln1_b = (const float*)d_in[2];
    const float* inw   = (const float*)d_in[3];
    const float* inb   = (const float*)d_in[4];
    const float* outw  = (const float*)d_in[5];
    const float* outb  = (const float*)d_in[6];
    const float* ln2_g = (const float*)d_in[7];
    const float* ln2_b = (const float*)d_in[8];
    const float* wg    = (const float*)d_in[9];
    const float* w1    = (const float*)d_in[10];
    const float* b1    = (const float*)d_in[11];
    const float* w2    = (const float*)d_in[12];
    const float* b2    = (const float*)d_in[13];
    float* out = (float*)d_out;
    (void)in_sizes; (void)n_in; (void)out_size; (void)ws_size;

    // workspace layout (bytes)
    char* base = (char*)d_ws;
    const size_t SM  = (size_t)SS * MM;
    __bf16* hA   = (__bf16*)base;                              // 16 MB: LN1/LN2 out
    __bf16* qkvb = (__bf16*)(base + SM*2);                     // 48 MB (region 64 MB)
    __bf16* h1b  = qkvb;                                       // 64 MB, reuses qkv region
    char*   r3   = base + SM*2 + (size_t)EE*CC*FF*2;
    __bf16* attn = (__bf16*)r3;                                // 16 MB
    float*  eo   = (float*)r3;                                 // 32 MB, reuses attn region
    float*  x1   = (float*)(r3 + (size_t)EE*CC*MM*4);          // 32 MB
    __bf16* disp = (__bf16*)((char*)x1 + SM*4);                // 16 MB
    __bf16* inwb = (__bf16*)((char*)disp + SM*2);              //  6 MB
    __bf16* outwb= (__bf16*)((char*)inwb + (size_t)3*MM*MM*2); //  2 MB
    __bf16* w1t  = (__bf16*)((char*)outwb + (size_t)MM*MM*2);  // 64 MB
    __bf16* w2t  = (__bf16*)((char*)w1t + (size_t)EE*MM*FF*2); // 64 MB
    int*    idxp = (int*)((char*)w2t + (size_t)EE*FF*MM*2);
    int*    posp = idxp + SS;
    float*  gval = (float*)(posp + SS);

    // 0) weight conversion: bf16 (+ transpose to N x K for the MoE weights)
    cvt_kernel<<<(unsigned)((size_t)3*MM*MM/1024), 256, 0, stream>>>(inw, inwb);
    cvt_kernel<<<(unsigned)((size_t)MM*MM/1024), 256, 0, stream>>>(outw, outwb);
    transpose_cvt_kernel<<<dim3(FF/32, MM/32, EE), 256, 0, stream>>>(w1, w1t, MM, FF);
    transpose_cvt_kernel<<<dim3(MM/32, FF/32, EE), 256, 0, stream>>>(w2, w2t, FF, MM);

    // 1) LN1 -> bf16
    ln_kernel<<<SS, 256, 0, stream>>>(x, ln1_g, ln1_b, hA);
    // 2) QKV = h @ in_proj_w^T + b   [S,3M] bf16
    gemm_kernel<<<dim3(3*MM/128, SS/128, 1), 256, 0, stream>>>(
        hA, inwb, inb, nullptr, qkvb, 1, 3*MM, MM, 0, 0, 0, 0, 0);
    // 3) causal flash attention -> [S,M] bf16
    attn_kernel<<<dim3(TT/128, HH, BB), 256, 0, stream>>>(qkvb, attn);
    // 4) x1 = x + attn @ out_proj_w^T + b   (f32)
    gemm_kernel<<<dim3(MM/128, SS/128, 1), 256, 0, stream>>>(
        attn, outwb, outb, x, x1, 0, MM, MM, 0, 0, 0, 0, 0);
    // 5) LN2 -> bf16
    ln_kernel<<<SS, 256, 0, stream>>>(x1, ln2_g, ln2_b, hA);
    // 6) gating
    gate_kernel<<<SS/8, 256, 0, stream>>>(hA, wg, idxp, gval);
    // 7) expert-queue positions (deterministic scan)
    scan_kernel<<<1, 256, 0, stream>>>(idxp, posp);
    // 8) zero dispatch buffer (16 MB bf16), then scatter kept tokens
    zero_kernel<<<(unsigned)(SM*2/16/256), 256, 0, stream>>>((float4*)disp);
    scatter_kernel<<<SS, 256, 0, stream>>>(hA, idxp, posp, disp);
    // 9) expert GEMM1: h1 = gelu(disp @ w1t^T + b1)   [E,C,4M] bf16
    gemm_kernel<<<dim3(FF/128, CC/128, EE), 256, 0, stream>>>(
        disp, w1t, b1, nullptr, h1b, 1, FF, MM, 1,
        (long long)CC*MM, (long long)MM*FF, (long long)FF, (long long)CC*FF);
    // 10) expert GEMM2: eo = h1 @ w2t^T + b2          [E,C,M] f32
    gemm_kernel<<<dim3(MM/128, CC/128, EE), 256, 0, stream>>>(
        h1b, w2t, b2, nullptr, eo, 0, MM, FF, 0,
        (long long)CC*FF, (long long)FF*MM, (long long)MM, (long long)CC*MM);
    // 11) combine: out = x1 + gate * eo[idx,pos]
    combine_kernel<<<SS, 256, 0, stream>>>(x1, eo, idxp, posp, gval, out);
}
